// Transformer_20864951124611
// MI455X (gfx1250) — matrix operand order repaired and checked by reading, not compiled, not run on hardware
//
#include <hip/hip_runtime.h>
#include <hip/hip_bf16.h>

// ---------------------------------------------------------------------------
// ViT-ish transformer forward for MI455X (gfx1250, wave32, WMMA).
// All heavy math runs through v_wmma_f32_16x16x32_f16 (f32 accumulate).
// Attention is flash-style (online softmax) so the 680MB logits tensor is
// never materialized; working set (~30MB) lives in L2.
// GEMM: one wave owns a 32x64 tile; B-fragments batched ahead of 8 WMMAs.
// Attention: 64-key chunks -> 8 WMMAs per LDS round-trip.
// ---------------------------------------------------------------------------

typedef __attribute__((ext_vector_type(16))) _Float16 v16h;
typedef __attribute__((ext_vector_type(8)))  float    v8f;

#define TOK 2304      // patches per image (48*48)
#define BSZ 2
#define DMODEL 256
#define NHEAD 8
#define DHEAD 32
#define DMLP 1024
#define MROWS (BSZ*TOK)   // 4608

union F16x16 { v16h v; uint4 q[2]; _Float16 h[16]; };

// A-operand fragment (16xK tile, 16-bit): lane holds row (lane&15);
// halves 0..7 -> K = base + (lane>=16?8:0) + 0..7,
// halves 8..15 -> K = base + 16 + (lane>=16?8:0) + 0..7.
__device__ inline v16h load_a_frag(const _Float16* __restrict__ base, int lane) {
  F16x16 f;
  const int lo = (lane & 16) ? 8 : 0;
  f.q[0] = *(const uint4*)(base + lo);
  f.q[1] = *(const uint4*)(base + lo + 16);
  return f.v;
}

// B-operand fragment (Kx16 tile, 16-bit, stored K-contiguous per column):
// lane holds column (lane&15); halves 0..7 -> K = base + (lane>=16?8:0) + 0..7,
// halves 8..15 -> K = base + 16 + (lane>=16?8:0) + 0..7.
__device__ inline v16h load_b_frag(const _Float16* __restrict__ base, int lane) {
  F16x16 f;
  const int lo = (lane & 16) ? 8 : 0;
  f.q[0] = *(const uint4*)(base + lo);
  f.q[1] = *(const uint4*)(base + lo + 16);
  return f.v;
}

__device__ inline v8f wmma_f16(v16h a, v16h b, v8f c) {
  return __builtin_amdgcn_wmma_f32_16x16x32_f16(
      /*neg_a=*/false, a, /*neg_b=*/false, b,
      /*c_mod=*/(short)0, c, /*reuse_a=*/false, /*reuse_b=*/false);
}

__device__ inline float gelu_exact(float x) {
  return 0.5f * x * (1.0f + erff(x * 0.70710678118654752f));
}

// ---------------------------------------------------------------------------
// im2col for 8x8/stride-8 patch conv: x[B,3,384,384] -> A0[4608,192] f16
// ---------------------------------------------------------------------------
__global__ void im2col_kernel(const float* __restrict__ x, _Float16* __restrict__ a0) {
  int i = blockIdx.x * 256 + threadIdx.x;
  if (i >= MROWS * 192) return;
  int row = i / 192, k = i % 192;
  int b = row / TOK, p = row % TOK;
  int py = p / 48, px = p % 48;
  int c = k / 64, rem = k % 64, iy = rem >> 3, ix = rem & 7;
  a0[i] = (_Float16)x[(((size_t)b * 3 + c) * 384 + py * 8 + iy) * 384 + px * 8 + ix];
}

// ---------------------------------------------------------------------------
// Pack f32 weights into column-major f16 Bt[n*K + k] so WMMA B-fragments are
// two contiguous 16B loads. mode 0: W[k*Nc+n]; 1: W[n*K+k]; 2: per-head QKV
// W[((n/32)*K + k)*32 + (n%32)].
// ---------------------------------------------------------------------------
__global__ void pack_b_kernel(const float* __restrict__ W, _Float16* __restrict__ Bt,
                              int K, int Nc, int mode) {
  int i = blockIdx.x * 256 + threadIdx.x;
  if (i >= K * Nc) return;
  int n = i / K, k = i % K;
  size_t idx;
  if (mode == 0)      idx = (size_t)k * Nc + n;
  else if (mode == 1) idx = (size_t)n * K + k;
  else                idx = ((size_t)(n >> 5) * K + k) * 32 + (n & 31);
  Bt[i] = (_Float16)W[idx];
}

// ---------------------------------------------------------------------------
// Generic WMMA GEMM: C[M,Nc] = act(A16[M,K] * Bt16[Nc,K]^T + bias[n]).
// One wave owns a 32x64 tile: 2 A-frags x 4 B-frags -> 8 WMMAs per k-step,
// with all B-frags loaded into registers before the WMMA burst so the
// scheduler can overlap the next load clause with matrix math.
// out_mode 0: row-major f32/f16 stores.  out_mode 1: QKV head-split f16
// store into [B, H, TOK, 32].  act: 0 none, 1 relu, 2 exact gelu.
// ---------------------------------------------------------------------------
__global__ void gemm_wmma_kernel(const _Float16* __restrict__ A,
                                 const _Float16* __restrict__ Bt,
                                 const float* __restrict__ bias,
                                 float* __restrict__ dst32,
                                 _Float16* __restrict__ dst16,
                                 int M, int K, int Nc, int act, int out_mode) {
  const int lane = threadIdx.x & 31;
  const int wave = threadIdx.x >> 5;
  const int nt64 = Nc >> 6;
  const int nunits = (M >> 5) * nt64;
  const int unit = blockIdx.x * 8 + wave;
  if (unit >= nunits) return;
  const int mtile = unit / nt64;
  const int ntile = unit % nt64;
  const int row0 = mtile << 5;
  const int col0 = ntile << 6;

  v8f acc[2][4] = {};
  const _Float16* arow0 = A + (size_t)(row0 + (lane & 15)) * K;
  const _Float16* arow1 = arow0 + (size_t)16 * K;
  const _Float16* bcol0 = Bt + (size_t)(col0 + (lane & 15)) * K;

  for (int kk = 0; kk < K; kk += 32) {
    __builtin_prefetch(arow0 + kk + 128, 0, 1);
    __builtin_prefetch(arow1 + kk + 128, 0, 1);
    v16h af0 = load_a_frag(arow0 + kk, lane);
    v16h af1 = load_a_frag(arow1 + kk, lane);
    v16h bf[4];
#pragma unroll
    for (int t = 0; t < 4; ++t)
      bf[t] = load_b_frag(bcol0 + (size_t)(t * 16) * K + kk, lane);
#pragma unroll
    for (int t = 0; t < 4; ++t)
      acc[0][t] = wmma_f16(af0, bf[t], acc[0][t]);
#pragma unroll
    for (int t = 0; t < 4; ++t)
      acc[1][t] = wmma_f16(af1, bf[t], acc[1][t]);
  }

  const int rlo = (lane & 16) ? 8 : 0;
#pragma unroll
  for (int mh = 0; mh < 2; ++mh) {
#pragma unroll
    for (int t = 0; t < 4; ++t) {
#pragma unroll
      for (int v = 0; v < 8; ++v) {
        int r = row0 + mh * 16 + v + rlo;
        int c = col0 + t * 16 + (lane & 15);
        float xv = acc[mh][t][v] + (bias ? bias[c] : 0.0f);
        if (act == 1)      xv = fmaxf(xv, 0.0f);
        else if (act == 2) xv = gelu_exact(xv);
        if (out_mode == 0) {
          if (dst32) dst32[(size_t)r * Nc + c] = xv;
          if (dst16) dst16[(size_t)r * Nc + c] = (_Float16)xv;
        } else {
          int b = r / TOK, nn = r % TOK, hh = c >> 5, e = c & 31;
          dst16[((((size_t)b * NHEAD + hh) * TOK + nn) << 5) + e] = (_Float16)xv;
        }
      }
    }
  }
}

// ---------------------------------------------------------------------------
// Flash attention (no 1/sqrt(d) scale, matching reference).
// Grid: (B*H, 18); block = 8 waves; wave w handles q-tile blockIdx.y*8+w.
// 64-key chunks staged in LDS: K as [key][dh], V transposed to [dh][key].
// S(16x64) = 4 WMMAs; online softmax with 16-lane shfl reductions; P goes
// through a per-wave LDS tile (C-layout -> A-layout); O += P*V = 4 WMMAs.
// ---------------------------------------------------------------------------
__global__ void attention_kernel(const _Float16* __restrict__ q16,
                                 const _Float16* __restrict__ k16,
                                 const _Float16* __restrict__ v16,
                                 float* __restrict__ out) {
  __shared__ _Float16 sK[64 * 32];      // [key][dh]
  __shared__ _Float16 sVt[32 * 64];     // [dh][key]
  __shared__ _Float16 sP[8][16 * 64];   // per-wave P tile [row][key]

  const int tid = threadIdx.x;
  const int lane = tid & 31;
  const int wave = tid >> 5;
  const int bh = blockIdx.x;            // 0..15
  const int b = bh >> 3, h = bh & 7;
  const int qtile = blockIdx.y * 8 + wave;  // 0..143

  const _Float16* qb = q16 + (size_t)bh * TOK * 32;
  const _Float16* kb = k16 + (size_t)bh * TOK * 32;
  const _Float16* vb = v16 + (size_t)bh * TOK * 32;

  const v16h qf = load_a_frag(qb + (size_t)(qtile * 16 + (lane & 15)) * 32, lane);

  float m[8], l[8];
  v8f o0 = {}, o1 = {};
#pragma unroll
  for (int i = 0; i < 8; ++i) { m[i] = -1e30f; l[i] = 0.0f; }

  for (int kc = 0; kc < TOK; kc += 64) {
    __syncthreads();
    {
      // stage K chunk: 64 keys x 32 dh  (contiguous copy, all 256 threads)
      int flat = tid * 8;
      *(uint4*)(sK + flat) = *(const uint4*)(kb + (size_t)kc * 32 + flat);
      // stage V chunk transposed: sVt[dh][key], key stride 64
      int key = flat >> 5, e0 = flat & 31;
      union { uint4 u; _Float16 hh[8]; } d;
      d.u = *(const uint4*)(vb + (size_t)(kc + key) * 32 + e0);
#pragma unroll
      for (int j = 0; j < 8; ++j) sVt[(e0 + j) * 64 + key] = d.hh[j];
    }
    __syncthreads();

    // S = Q * K^T  (four 16-key tiles)
    v8f s[4];
#pragma unroll
    for (int t = 0; t < 4; ++t) {
      v16h kf = load_b_frag(sK + ((lane & 15) + t * 16) * 32, lane);
      v8f z = {};
      s[t] = wmma_f16(qf, kf, z);
    }

    _Float16* myP = sP[wave];
    const int rlo = (lane & 16) ? 8 : 0;
#pragma unroll
    for (int v = 0; v < 8; ++v) {
      float a0 = s[0][v], a1 = s[1][v], a2 = s[2][v], a3 = s[3][v];
      float mx = fmaxf(fmaxf(a0, a1), fmaxf(a2, a3));
#pragma unroll
      for (int off = 1; off < 16; off <<= 1) mx = fmaxf(mx, __shfl_xor(mx, off, 32));
      float mn = fmaxf(m[v], mx);
      float sc = __expf(m[v] - mn);
      float p0 = __expf(a0 - mn);
      float p1 = __expf(a1 - mn);
      float p2 = __expf(a2 - mn);
      float p3 = __expf(a3 - mn);
      float ps = (p0 + p1) + (p2 + p3);
#pragma unroll
      for (int off = 1; off < 16; off <<= 1) ps += __shfl_xor(ps, off, 32);
      l[v] = l[v] * sc + ps;
      m[v] = mn;
      o0[v] *= sc;
      o1[v] *= sc;
      int r = v + rlo, c = lane & 15;
      myP[r * 64 + c]      = (_Float16)p0;
      myP[r * 64 + c + 16] = (_Float16)p1;
      myP[r * 64 + c + 32] = (_Float16)p2;
      myP[r * 64 + c + 48] = (_Float16)p3;
    }

    // O += P * V : P is 16x64 (two A k-steps), V is 64x32 (two dh tiles)
    v16h pf0 = load_a_frag(myP + (lane & 15) * 64, lane);
    v16h pf1 = load_a_frag(myP + (lane & 15) * 64 + 32, lane);
    v16h vf00 = load_b_frag(sVt + (lane & 15) * 64, lane);
    v16h vf01 = load_b_frag(sVt + (lane & 15) * 64 + 32, lane);
    v16h vf10 = load_b_frag(sVt + ((lane & 15) + 16) * 64, lane);
    v16h vf11 = load_b_frag(sVt + ((lane & 15) + 16) * 64 + 32, lane);
    o0 = wmma_f16(pf0, vf00, o0);
    o0 = wmma_f16(pf1, vf01, o0);
    o1 = wmma_f16(pf0, vf10, o1);
    o1 = wmma_f16(pf1, vf11, o1);
  }

  const int rlo = (lane & 16) ? 8 : 0;
#pragma unroll
  for (int v = 0; v < 8; ++v) {
    int r = qtile * 16 + v + rlo;
    float inv = 1.0f / l[v];
    size_t base = ((size_t)b * TOK + r) * DMODEL + h * DHEAD;
    out[base + (lane & 15)]      = o0[v] * inv;
    out[base + 16 + (lane & 15)] = o1[v] * inv;
  }
}

// ---------------------------------------------------------------------------
// dst = LayerNorm(x + res) * w + b  over D=256; optional f16 mirror.
// One 256-thread block per row.
// ---------------------------------------------------------------------------
__global__ void add_ln_kernel(const float* __restrict__ x, const float* __restrict__ res,
                              const float* __restrict__ w, const float* __restrict__ bia,
                              float* __restrict__ dst32, _Float16* __restrict__ dst16) {
  __shared__ float red[256];
  const int row = blockIdx.x;
  const int t = threadIdx.x;
  const size_t idx = (size_t)row * DMODEL + t;
  float v = x[idx] + (res ? res[idx] : 0.0f);

  red[t] = v; __syncthreads();
  for (int s = 128; s > 0; s >>= 1) { if (t < s) red[t] += red[t + s]; __syncthreads(); }
  float mu = red[0] * (1.0f / DMODEL);
  __syncthreads();

  float d = v - mu;
  red[t] = d * d; __syncthreads();
  for (int s = 128; s > 0; s >>= 1) { if (t < s) red[t] += red[t + s]; __syncthreads(); }
  float var = red[0] * (1.0f / DMODEL);

  float y = d * rsqrtf(var + 1e-6f) * w[t] + bia[t];
  dst32[idx] = y;
  if (dst16) dst16[idx] = (_Float16)y;
}

// ---------------------------------------------------------------------------
// Host-side orchestration
// ---------------------------------------------------------------------------
extern "C" void kernel_launch(void* const* d_in, const int* in_sizes, int n_in,
                              void* d_out, int out_size, void* d_ws, size_t ws_size,
                              hipStream_t stream) {
  const float* x      = (const float*)d_in[0];
  const float* conv_w = (const float*)d_in[1];
  const float* conv_b = (const float*)d_in[2];
  const float* Wq     = (const float*)d_in[3];
  const float* bq     = (const float*)d_in[4];
  const float* Wk     = (const float*)d_in[5];
  const float* bk     = (const float*)d_in[6];
  const float* Wv     = (const float*)d_in[7];
  const float* bv     = (const float*)d_in[8];
  const float* ln1_w  = (const float*)d_in[9];
  const float* ln1_b  = (const float*)d_in[10];
  const float* fc1_w  = (const float*)d_in[11];
  const float* fc1_b  = (const float*)d_in[12];
  const float* fc2_w  = (const float*)d_in[13];
  const float* fc2_b  = (const float*)d_in[14];
  const float* lnf_w  = (const float*)d_in[15];
  const float* lnf_b  = (const float*)d_in[16];

  char* ws = (char*)d_ws;
  size_t off = 0;
  auto carve = [&](size_t bytes) -> void* {
    void* p = ws + off;
    off = (off + bytes + 255) & ~(size_t)255;
    return p;
  };
  float*    h32   = (float*)   carve((size_t)MROWS * DMODEL * 4);
  float*    tmp32 = (float*)   carve((size_t)MROWS * DMODEL * 4);   // attn out / mlp out
  _Float16* hf16  = (_Float16*)carve((size_t)MROWS * DMODEL * 2);
  _Float16* mf16  = (_Float16*)carve((size_t)MROWS * DMLP   * 2);
  _Float16* q16   = (_Float16*)carve((size_t)BSZ * NHEAD * TOK * DHEAD * 2);
  _Float16* k16   = (_Float16*)carve((size_t)BSZ * NHEAD * TOK * DHEAD * 2);
  _Float16* v16p  = (_Float16*)carve((size_t)BSZ * NHEAD * TOK * DHEAD * 2);
  _Float16* a016  = (_Float16*)carve((size_t)MROWS * 192 * 2);
  _Float16* bt16  = (_Float16*)carve((size_t)DMLP * DMODEL * 2);    // packed B, reused

  auto gemm_grid = [](int M, int Nc) { return ((M >> 5) * (Nc >> 6) + 7) / 8; };
  auto pack_grid = [](int K, int Nc) { return (K * Nc + 255) / 256; };

  // --- Patch embedding: h = im2col(x) @ conv_w^T + conv_b ---
  im2col_kernel<<<(MROWS * 192 + 255) / 256, 256, 0, stream>>>(x, a016);
  pack_b_kernel<<<pack_grid(192, DMODEL), 256, 0, stream>>>(conv_w, bt16, 192, DMODEL, 1);
  gemm_wmma_kernel<<<gemm_grid(MROWS, DMODEL), 256, 0, stream>>>(
      a016, bt16, conv_b, h32, hf16, MROWS, 192, DMODEL, /*act*/0, /*out_mode*/0);

  for (int l = 0; l < 2; ++l) {
    const float* Ws[3] = { Wq + (size_t)l * NHEAD * DMODEL * DHEAD,
                           Wk + (size_t)l * NHEAD * DMODEL * DHEAD,
                           Wv + (size_t)l * NHEAD * DMODEL * DHEAD };
    const float* bs[3] = { bq + l * DMODEL, bk + l * DMODEL, bv + l * DMODEL };
    _Float16*    ds[3] = { q16, k16, v16p };

    // QKV: relu(h @ W + b), per-head, f16 output [B,H,N,32]
    for (int j = 0; j < 3; ++j) {
      pack_b_kernel<<<pack_grid(DMODEL, DMODEL), 256, 0, stream>>>(
          Ws[j], bt16, DMODEL, DMODEL, 2);
      gemm_wmma_kernel<<<gemm_grid(MROWS, DMODEL), 256, 0, stream>>>(
          hf16, bt16, bs[j], nullptr, ds[j], MROWS, DMODEL, DMODEL, /*relu*/1, /*qkv*/1);
    }

    // Flash attention -> tmp32 [B*N, 256]
    attention_kernel<<<dim3(BSZ * NHEAD, TOK / 16 / 8), 256, 0, stream>>>(
        q16, k16, v16p, tmp32);

    // h = LN(h + attn)
    add_ln_kernel<<<MROWS, 256, 0, stream>>>(
        h32, tmp32, ln1_w + l * DMODEL, ln1_b + l * DMODEL, h32, hf16);

    // MLP: gelu(h @ fc1 + b1) -> gelu(m @ fc2 + b2)
    pack_b_kernel<<<pack_grid(DMODEL, DMLP), 256, 0, stream>>>(
        fc1_w + (size_t)l * DMODEL * DMLP, bt16, DMODEL, DMLP, 0);
    gemm_wmma_kernel<<<gemm_grid(MROWS, DMLP), 256, 0, stream>>>(
        hf16, bt16, fc1_b + l * DMLP, nullptr, mf16, MROWS, DMODEL, DMLP, /*gelu*/2, 0);

    pack_b_kernel<<<pack_grid(DMLP, DMODEL), 256, 0, stream>>>(
        fc2_w + (size_t)l * DMLP * DMODEL, bt16, DMLP, DMODEL, 0);
    gemm_wmma_kernel<<<gemm_grid(MROWS, DMODEL), 256, 0, stream>>>(
        mf16, bt16, fc2_b + l * DMODEL, tmp32, nullptr, MROWS, DMLP, DMODEL, /*gelu*/2, 0);

    // h = LN(h + m)   (reference reuses ln1)
    add_ln_kernel<<<MROWS, 256, 0, stream>>>(
        h32, tmp32, ln1_w + l * DMODEL, ln1_b + l * DMODEL, h32, hf16);
  }

  // Final LayerNorm -> d_out
  add_ln_kernel<<<MROWS, 256, 0, stream>>>(
      h32, nullptr, lnf_w, lnf_b, (float*)d_out, nullptr);
}
